// SentenceLevelEncoder_6777458393260
// MI455X (gfx1250) — matrix operand order repaired and checked by reading, not compiled, not run on hardware
//
#include <hip/hip_runtime.h>
#include <hip/hip_bf16.h>

typedef _Float16 h16;
typedef __attribute__((ext_vector_type(16))) _Float16 v16h;
typedef __attribute__((ext_vector_type(8)))  _Float16 v8h;
typedef __attribute__((ext_vector_type(8)))  float    v8f;

#define BB 32
#define TT 32
#define HH 512
#define HALF 256
#define VDIM 2048
#define SDIM 768
#define NOBJ 36
#define NVP  20

// ---------------- WMMA fragment loaders ----------------
// A-matrix 16x32 f16: lane r (0-15): K 0-7 then 16-23; lane r+16: K 8-15 then 24-31.
__device__ __forceinline__ v16h load_a_frag(const h16* A, int lda, int m0, int k0,
                                            int r, int hh) {
  const h16* p = A + (size_t)(m0 + r) * lda + k0 + hh * 8;
  v8h lo = *(const v8h*)p;        // K = k0+hh*8   .. +7
  v8h hi = *(const v8h*)(p + 16); // K = k0+16+hh*8.. +7
  return __builtin_shufflevector(lo, hi, 0,1,2,3,4,5,6,7,8,9,10,11,12,13,14,15);
}
// B-matrix 32x16 f16 for x@W.T: B[k][n] = W[n][k]. lane n: K 0-7 then 16-23 (lanes 0-15) or 8-15 then 24-31.
__device__ __forceinline__ v16h load_b_frag(const h16* W, int ldw, int n0, int k0,
                                            int r, int hh) {
  const h16* p = W + (size_t)(n0 + r) * ldw + k0 + hh * 8;
  v8h lo = *(const v8h*)p;
  v8h hi = *(const v8h*)(p + 16);
  return __builtin_shufflevector(lo, hi, 0,1,2,3,4,5,6,7,8,9,10,11,12,13,14,15);
}

__device__ __forceinline__ float sigf(float x) { return 1.0f / (1.0f + expf(-x)); }

// ---------------- generic WMMA GEMM: C[M,N] = A[M,K] @ W[N,K]^T + bias ----------------
__global__ void gemm_wmma_kernel(const h16* __restrict__ A, const h16* __restrict__ W,
                                 const float* __restrict__ bias,
                                 float* __restrict__ C, h16* __restrict__ Ch,
                                 int M, int N, int K) {
  int wave = threadIdx.x >> 5;
  int gw = blockIdx.x * (blockDim.x >> 5) + wave;
  int tilesN = N >> 4;
  int tilesM = M >> 4;
  if (gw >= tilesM * tilesN) return;      // wave-uniform: EXEC all ones inside
  int tm = gw / tilesN, tn = gw % tilesN;
  int lane = threadIdx.x & 31;
  int r = lane & 15, hh = lane >> 4;
  int col = tn * 16 + r;

  float binit = bias ? bias[col] : 0.0f;
  v8f c;
  for (int v = 0; v < 8; ++v) c[v] = binit;

  for (int k0 = 0; k0 < K; k0 += 32) {
    v16h a = load_a_frag(A, K, tm * 16, k0, r, hh);
    v16h b = load_b_frag(W, K, tn * 16, k0, r, hh);
    c = __builtin_amdgcn_wmma_f32_16x16x32_f16(false, a, false, b, (short)0, c,
                                               false, false);
  }
  for (int v = 0; v < 8; ++v) {
    int row = tm * 16 + hh * 8 + v;
    float val = c[v];
    C[(size_t)row * N + col] = val;
    if (Ch) Ch[(size_t)row * N + col] = (h16)val;
  }
}

// ---------------- f32 -> f16 convert ----------------
__global__ void cvt_kernel(const float* __restrict__ in, h16* __restrict__ out, int n) {
  for (int i = blockIdx.x * blockDim.x + threadIdx.x; i < n; i += gridDim.x * blockDim.x)
    out[i] = (h16)in[i];
}

__global__ void addvec_kernel(const float* __restrict__ a, const float* __restrict__ b,
                              float* __restrict__ o, int n) {
  int i = blockIdx.x * blockDim.x + threadIdx.x;
  if (i < n) o[i] = a[i] + b[i];
}

// ---------------- additive attention: one block per (b,t) ----------------
__global__ void attn_kernel(const float* __restrict__ wf2d, const float* __restrict__ u,
                            const float* __restrict__ bias, const float* __restrict__ ww,
                            float* __restrict__ outp, int N) {
  __shared__ float red[256];
  __shared__ float scores[40];
  int bt = blockIdx.x;           // b*T + t
  int b = bt >> 5;               // / T
  int tid = threadIdx.x;
  const float* wf = wf2d + (size_t)bt * HH;
  float wf0 = wf[tid], wf1 = wf[tid + 256];
  float bi0 = bias[tid], bi1 = bias[tid + 256];
  float w0 = ww[tid], w1 = ww[tid + 256];

  for (int n = 0; n < N; ++n) {
    const float* up = u + (size_t)(b * N + n) * HH;
    float f0 = wf0 + up[tid] + bi0;
    float f1 = wf1 + up[tid + 256] + bi1;
    red[tid] = tanhf(f0) * w0 + tanhf(f1) * w1;
    __syncthreads();
    for (int s = 128; s > 0; s >>= 1) {
      if (tid < s) red[tid] += red[tid + s];
      __syncthreads();
    }
    if (tid == 0) scores[n] = red[0];
    __syncthreads();
  }
  // softmax over N (constant wb cancels in softmax)
  float mx = -1e30f;
  for (int n = 0; n < N; ++n) mx = fmaxf(mx, scores[n]);
  float sm = 0.0f;
  for (int n = 0; n < N; ++n) sm += expf(scores[n] - mx);
  float inv = 1.0f / sm;
  float a0 = 0.0f, a1 = 0.0f;
  for (int n = 0; n < N; ++n) {
    float aw = expf(scores[n] - mx) * inv;
    const float* up = u + (size_t)(b * N + n) * HH;
    a0 += aw * (wf0 + up[tid] + bi0);
    a1 += aw * (wf1 + up[tid + 256] + bi1);
  }
  outp[(size_t)bt * HH + tid] = a0;
  outp[(size_t)bt * HH + tid + 256] = a1;
}

// ---------------- concat [f2d | attn_motion | attn_obj] -> f16 [1024,1536] ----------------
__global__ void concat_kernel(const float* __restrict__ f2d, const float* __restrict__ am,
                              const float* __restrict__ ao, h16* __restrict__ feat16) {
  int i = blockIdx.x * 256 + threadIdx.x;   // 1024*512
  int row = i >> 9, col = i & 511;
  size_t o = (size_t)row * 1536 + col;
  feat16[o]        = (h16)f2d[i];
  feat16[o + 512]  = (h16)am[i];
  feat16[o + 1024] = (h16)ao[i];
}

// ---------------- persistent bidirectional LSTM: block 0 = fwd, block 1 = bwd ----------------
// dyn LDS: gates 32x1024 f32 (128KB) | c 32x256 f32 (32KB) | h16 32x256 (16KB)
__global__ void lstm_kernel(const float* __restrict__ gatesF, const float* __restrict__ gatesB,
                            const h16* __restrict__ whhF, const h16* __restrict__ whhB,
                            float* __restrict__ out) {
  extern __shared__ char smem[];
  float* g   = (float*)smem;                                  // [32][1024]
  float* c   = (float*)(smem + 32 * 1024 * 4);                // [32][256]
  h16*  hbuf = (h16*) (smem + 32 * 1024 * 4 + 32 * 256 * 4);  // [32][256]

  int d = blockIdx.x;
  const float* gatesIn = d ? gatesB : gatesF;
  const h16* whh = d ? whhB : whhF;
  int tid = threadIdx.x;

  for (int i = tid; i < 32 * 256; i += 1024) { c[i] = 0.0f; hbuf[i] = (h16)0.0f; }
  __syncthreads();

  int wave = tid >> 5, lane = tid & 31, r = lane & 15, hh = lane >> 4;

  for (int t = 0; t < TT; ++t) {
    int tt = d ? (TT - 1 - t) : t;
    // g = gates_in[:,tt,:] + h @ whh^T   (M=32, N=1024, K=256; 128 tiles / 32 waves)
    for (int q = 0; q < 4; ++q) {
      int tile = wave * 4 + q;
      int tm = tile >> 6, tn = tile & 63;
      int col = tn * 16 + r;
      v8f acc;
      for (int v = 0; v < 8; ++v) {
        int b = tm * 16 + hh * 8 + v;
        acc[v] = gatesIn[(size_t)(b * TT + tt) * 1024 + col];
      }
      for (int k0 = 0; k0 < HALF; k0 += 32) {
        v16h a = load_a_frag(hbuf, HALF, tm * 16, k0, r, hh);
        v16h bf = load_b_frag(whh, HALF, tn * 16, k0, r, hh);
        acc = __builtin_amdgcn_wmma_f32_16x16x32_f16(false, a, false, bf, (short)0, acc,
                                                     false, false);
      }
      for (int v = 0; v < 8; ++v) {
        int b = tm * 16 + hh * 8 + v;
        g[b * 1024 + col] = acc[v];
      }
    }
    __syncthreads();
    // elementwise gate math: 32*256 cells, 8 per thread
    for (int q = 0; q < 8; ++q) {
      int idx = tid + q * 1024;
      int b = idx >> 8, j = idx & 255;
      const float* gb = g + b * 1024;
      float gi = gb[j], gf = gb[256 + j], gg = gb[512 + j], go = gb[768 + j];
      float cv = sigf(gf) * c[idx] + sigf(gi) * tanhf(gg);
      float hv = sigf(go) * tanhf(cv);
      c[idx] = cv;
      hbuf[idx] = (h16)hv;
      out[(size_t)(b * TT + tt) * HH + d * HALF + j] = hv;
    }
    __syncthreads();
  }
}

// ---------------- max over T ----------------
__global__ void maxpool_kernel(const float* __restrict__ out, float* __restrict__ video) {
  int i = blockIdx.x * 256 + threadIdx.x;   // 32*512
  int b = i >> 9, j = i & 511;
  float m = -1e30f;
  for (int t = 0; t < TT; ++t) m = fmaxf(m, out[(size_t)(b * TT + t) * HH + j]);
  video[i] = m;
}

// ---------------- final FC: [32,512] @ [768,512]^T ----------------
__global__ void fc_kernel(const float* __restrict__ video, const float* __restrict__ w,
                          const float* __restrict__ bias, float* __restrict__ outp) {
  int i = blockIdx.x * 256 + threadIdx.x;   // 32*768
  int b = i / SDIM, s = i % SDIM;
  float acc = bias[s];
  const float* v = video + b * HH;
  const float* wr = w + (size_t)s * HH;
  for (int k = 0; k < HH; ++k) acc += v[k] * wr[k];
  outp[i] = acc;
}

// =======================================================================
extern "C" void kernel_launch(void* const* d_in, const int* in_sizes, int n_in,
                              void* d_out, int out_size, void* d_ws, size_t ws_size,
                              hipStream_t stream) {
  const float* visual = (const float*)d_in[0];
  const float* vp     = (const float*)d_in[1];
  const float* obj    = (const float*)d_in[2];
  const float* w2d_w  = (const float*)d_in[3];
  const float* w2d_b  = (const float*)d_in[4];
  const float* W_w    = (const float*)d_in[5];
  const float* W_b    = (const float*)d_in[6];
  const float* Uo_w   = (const float*)d_in[7];
  const float* Uo_b   = (const float*)d_in[8];
  const float* Um_w   = (const float*)d_in[9];
  const float* Um_b   = (const float*)d_in[10];
  const float* bo     = (const float*)d_in[11];
  const float* bm     = (const float*)d_in[12];
  const float* wo_w   = (const float*)d_in[13];
  const float* wm_w   = (const float*)d_in[15];
  const float* wih_f  = (const float*)d_in[17];
  const float* whh_f  = (const float*)d_in[18];
  const float* bih_f  = (const float*)d_in[19];
  const float* bhh_f  = (const float*)d_in[20];
  const float* wih_b  = (const float*)d_in[21];
  const float* whh_b  = (const float*)d_in[22];
  const float* bih_b  = (const float*)d_in[23];
  const float* bhh_b  = (const float*)d_in[24];
  const float* fc_w   = (const float*)d_in[25];
  const float* fc_b   = (const float*)d_in[26];

  float* out_seq = (float*)d_out;                       // [32,32,512]
  float* out_sem = (float*)d_out + BB * TT * HH;        // [32,768]

  // scratch bump allocator
  char* ws = (char*)d_ws;
  size_t off = 0;
  auto alloc = [&](size_t bytes) -> void* {
    void* p = ws + off;
    off = (off + bytes + 255) & ~(size_t)255;
    return p;
  };
  const int MT = BB * TT;   // 1024
  h16* vis16   = (h16*)alloc((size_t)MT * VDIM * 2);
  h16* w2d16   = (h16*)alloc((size_t)HH * VDIM * 2);
  h16* Ww16    = (h16*)alloc((size_t)HH * HH * 2);
  h16* Uo16    = (h16*)alloc((size_t)HH * HH * 2);
  h16* Um16    = (h16*)alloc((size_t)HH * HH * 2);
  h16* obj16   = (h16*)alloc((size_t)BB * NOBJ * HH * 2);
  h16* vp16    = (h16*)alloc((size_t)BB * NVP * HH * 2);
  h16* wihF16  = (h16*)alloc((size_t)1024 * 1536 * 2);
  h16* wihB16  = (h16*)alloc((size_t)1024 * 1536 * 2);
  h16* whhF16  = (h16*)alloc((size_t)1024 * HALF * 2);
  h16* whhB16  = (h16*)alloc((size_t)1024 * HALF * 2);
  float* biasF = (float*)alloc(1024 * 4);
  float* biasB = (float*)alloc(1024 * 4);
  float* f2d   = (float*)alloc((size_t)MT * HH * 4);
  h16*   f2d16 = (h16*)  alloc((size_t)MT * HH * 2);
  float* wf2d  = (float*)alloc((size_t)MT * HH * 4);
  float* uobj  = (float*)alloc((size_t)BB * NOBJ * HH * 4);
  float* umot  = (float*)alloc((size_t)BB * NVP * HH * 4);
  float* attnO = (float*)alloc((size_t)MT * HH * 4);
  float* attnM = (float*)alloc((size_t)MT * HH * 4);
  h16*   feat16= (h16*)  alloc((size_t)MT * 1536 * 2);
  float* gatesF= (float*)alloc((size_t)MT * 1024 * 4);
  float* gatesB= (float*)alloc((size_t)MT * 1024 * 4);
  float* video = (float*)alloc((size_t)BB * HH * 4);

  auto cvt = [&](const float* src, h16* dst, int n) {
    int blocks = (n + 255) / 256; if (blocks > 2048) blocks = 2048;
    cvt_kernel<<<blocks, 256, 0, stream>>>(src, dst, n);
  };
  cvt(visual, vis16, MT * VDIM);
  cvt(w2d_w, w2d16, HH * VDIM);
  cvt(W_w, Ww16, HH * HH);
  cvt(Uo_w, Uo16, HH * HH);
  cvt(Um_w, Um16, HH * HH);
  cvt(obj, obj16, BB * NOBJ * HH);
  cvt(vp, vp16, BB * NVP * HH);
  cvt(wih_f, wihF16, 1024 * 1536);
  cvt(wih_b, wihB16, 1024 * 1536);
  cvt(whh_f, whhF16, 1024 * HALF);
  cvt(whh_b, whhB16, 1024 * HALF);
  addvec_kernel<<<4, 256, 0, stream>>>(bih_f, bhh_f, biasF, 1024);
  addvec_kernel<<<4, 256, 0, stream>>>(bih_b, bhh_b, biasB, 1024);

  auto gemm = [&](const h16* A, const h16* W, const float* bias, float* C, h16* Ch,
                  int M, int N, int K) {
    int tiles = (M >> 4) * (N >> 4);
    int blocks = (tiles + 7) / 8;
    gemm_wmma_kernel<<<blocks, 256, 0, stream>>>(A, W, bias, C, Ch, M, N, K);
  };
  // feature2ds = visual @ w2d^T + b      [1024,512]  (also f16 mirror)
  gemm(vis16, w2d16, w2d_b, f2d, f2d16, MT, HH, VDIM);
  // w_f2d = feature2ds @ W^T + b
  gemm(f2d16, Ww16, W_b, wf2d, nullptr, MT, HH, HH);
  // u_objs / u_motion
  gemm(obj16, Uo16, Uo_b, uobj, nullptr, BB * NOBJ, HH, HH);
  gemm(vp16, Um16, Um_b, umot, nullptr, BB * NVP, HH, HH);

  // additive attention (softmax over N)
  attn_kernel<<<MT, 256, 0, stream>>>(wf2d, uobj, bo, wo_w, attnO, NOBJ);
  attn_kernel<<<MT, 256, 0, stream>>>(wf2d, umot, bm, wm_w, attnM, NVP);

  // features = [f2d | attn_motion | attn_objects] -> f16
  concat_kernel<<<MT * HH / 256, 256, 0, stream>>>(f2d, attnM, attnO, feat16);

  // LSTM input projections (time-invariant) for both directions
  gemm(feat16, wihF16, biasF, gatesF, nullptr, MT, 1024, 1536);
  gemm(feat16, wihB16, biasB, gatesB, nullptr, MT, 1024, 1536);

  // persistent recurrence (block 0 fwd, block 1 bwd); 176KB dynamic LDS
  size_t lds = 32 * 1024 * 4 + 32 * 256 * 4 + 32 * 256 * 2;
  lstm_kernel<<<2, 1024, lds, stream>>>(gatesF, gatesB, whhF16, whhB16, out_seq);

  // head
  maxpool_kernel<<<BB * HH / 256, 256, 0, stream>>>(out_seq, video);
  fc_kernel<<<BB * SDIM / 256, 256, 0, stream>>>(video, fc_w, fc_b, out_sem);
}
